// Plant_MAE_SEG_76227079570101
// MI455X (gfx1250) — matrix operand order repaired and checked by reading, not compiled, not run on hardware
//
#include <hip/hip_runtime.h>
#include <cstdint>
#include <cstddef>

// ---------------------------------------------------------------------------
// Plant-MAE segmentation forward for MI455X (gfx1250, wave32, WMMA).
//
// Input pointer layout assumption (documented, unverifiable in compile-only):
//   d_in[0] = pts (B,3,N) f32, d_in[1] = cls_label (B,10) f32, then the
//   'params' pytree flattened in JAX tree order (dict keys sorted, lists in
//   index order).  Leaf index table is computed in kernel_launch.
//
// All dense layers go through k_gemm_wmma: f32 activations/weights staged to
// LDS as f16 in *fragment order* (operand reads are ds_load_b128, staging
// writes are ds_store_b128/b64), double-buffered with register staging,
// V_WMMA_F32_16X16X32_F16 f32 accumulation, fully unrolled epilogue
// (bias / residual / BN-affine / relu|gelu|leaky) with no scratch spill.
// ---------------------------------------------------------------------------

typedef __attribute__((ext_vector_type(16))) _Float16 v16h;
typedef __attribute__((ext_vector_type(8)))  _Float16 v8h;
typedef __attribute__((ext_vector_type(4)))  _Float16 v4h;
typedef __attribute__((ext_vector_type(8)))  float    v8f;

// ---------------- generic WMMA GEMM:  C = epi(A[M,K] @ W[K,N]) -------------
// Block tile 128x64, K-step 32, 8 waves (256 thr), each wave 32x32 (2x2 WMMA).
// LDS layouts are pre-swizzled so every lane's 16 operand halves are
// contiguous (two ds_load_b128 per fragment):
//   As[buf][row*40 + k]         (k contiguous: lane half h reads k = 16h..16h+15)
//   Bs[buf][col*40 + k]         (B stored column-major in LDS)
// Staging is chunked so each thread's LDS writes are contiguous:
//   A: 8 consecutive K in one aligned group -> one ds_store_b128
//   B: 4 consecutive K in one column       -> one ds_store_b64
__global__ __launch_bounds__(256) void k_gemm_wmma(
    const float* __restrict__ A, const float* __restrict__ W,
    const float* __restrict__ bias, const float* __restrict__ scale,
    const float* __restrict__ shift, const float* __restrict__ res,
    float* __restrict__ C, int M, int N, int K, int ldc, int colOff,
    int act, int resDiv)
{
  __shared__ _Float16 As[2][128 * 40];
  __shared__ _Float16 Bs[2][64 * 40];

  const int tid  = threadIdx.x;
  const int wid  = tid >> 5, lane = tid & 31;
  const int wm   = (wid & 3) * 32;       // 4 waves along M
  const int wn   = (wid >> 2) * 32;      // 2 waves along N
  const int hl   = lane >> 4, lr = lane & 15;
  const int bm   = blockIdx.y * 128, bn = blockIdx.x * 64;

  const v8f vz = {0.f,0.f,0.f,0.f,0.f,0.f,0.f,0.f};
  v8f acc[2][2];
  acc[0][0] = vz; acc[0][1] = vz; acc[1][0] = vz; acc[1][1] = vz;

  float ra[16];   // A slice: 2 chunks of 8 consecutive K   (128*32 / 256 thr)
  float rb[8];    // B slice: 2 quads  of 4 consecutive K   ( 32*64 / 256 thr)

  const int kt = (K + 31) >> 5;

  // A chunk c (0..511): row = c>>2, grp = c&3 (8-aligned K group)
  // B quad  q (0..511): kq = q>>6, col = q&63
  auto loadTile = [&](int t) {
    const int k0 = t << 5;
    #pragma unroll
    for (int j = 0; j < 2; ++j) {
      int ch = tid + j * 256;
      int row = ch >> 2, grp = ch & 3;
      int gm = bm + row;
      const float* Ap = A + (size_t)gm * K + k0 + grp * 8;
      #pragma unroll
      for (int i = 0; i < 8; ++i) {
        int gk = k0 + grp * 8 + i;
        ra[j * 8 + i] = (gm < M && gk < K) ? Ap[i] : 0.f;
      }
    }
    #pragma unroll
    for (int j = 0; j < 2; ++j) {
      int qd = tid + j * 256;
      int kq = qd >> 6, col = qd & 63;
      int gn = bn + col;
      #pragma unroll
      for (int i = 0; i < 4; ++i) {
        int gk = k0 + kq * 4 + i;
        rb[j * 4 + i] = (gk < K && gn < N) ? W[(size_t)gk * N + gn] : 0.f;
      }
    }
  };
  auto storeTile = [&](int s) {
    #pragma unroll
    for (int j = 0; j < 2; ++j) {
      int ch = tid + j * 256;
      int row = ch >> 2, grp = ch & 3;
      int slotbase = grp << 3;                               // 0,8,16,24
      v8h v;
      #pragma unroll
      for (int i = 0; i < 8; ++i) v[i] = (_Float16)ra[j * 8 + i];
      *(v8h*)(&As[s][row * 40 + slotbase]) = v;              // ds_store_b128
    }
    #pragma unroll
    for (int j = 0; j < 2; ++j) {
      int qd = tid + j * 256;
      int kq = qd >> 6, col = qd & 63;
      v4h v;
      #pragma unroll
      for (int i = 0; i < 4; ++i) v[i] = (_Float16)rb[j * 4 + i];
      *(v4h*)(&Bs[s][col * 40 + kq * 4]) = v;                // ds_store_b64
    }
  };

  loadTile(0);
  storeTile(0);
  __syncthreads();

  for (int t = 0; t < kt; ++t) {
    const int s = t & 1;
    if (t + 1 < kt) loadTile(t + 1);           // prefetch next tile into regs

    union Frag { v16h v; v8h p[2]; } af[2], bf[2];
    #pragma unroll
    for (int mi = 0; mi < 2; ++mi) {
      const _Float16* p = &As[s][(wm + mi * 16 + lr) * 40 + hl * 16];
      af[mi].p[0] = *(const v8h*)(p);
      af[mi].p[1] = *(const v8h*)(p + 8);
    }
    #pragma unroll
    for (int ni = 0; ni < 2; ++ni) {
      const _Float16* p = &Bs[s][(wn + ni * 16 + lr) * 40 + hl * 16];
      bf[ni].p[0] = *(const v8h*)(p);
      bf[ni].p[1] = *(const v8h*)(p + 8);
    }
    #pragma unroll
    for (int mi = 0; mi < 2; ++mi)
      #pragma unroll
      for (int ni = 0; ni < 2; ++ni)
        acc[mi][ni] = __builtin_amdgcn_wmma_f32_16x16x32_f16(
            false, af[mi].v, false, bf[ni].v, (short)0, acc[mi][ni], false, false);

    if (t + 1 < kt) storeTile((t + 1) & 1);    // fill the other buffer
    __syncthreads();
  }

  // epilogue (fully unrolled so acc stays in VGPRs):
  // C/D layout: VGPR r -> M = hl*8 + r, N = lane&15
  #pragma unroll
  for (int mi = 0; mi < 2; ++mi)
    #pragma unroll
    for (int ni = 0; ni < 2; ++ni) {
      const int n = bn + wn + ni * 16 + lr;
      if (n >= N) continue;
      const float bsv = bias  ? bias[n]  : 0.f;
      const float scv = scale ? scale[n] : 1.f;
      const float shv = scale ? shift[n] : 0.f;
      const int mbase = bm + wm + mi * 16 + hl * 8;
      #pragma unroll
      for (int r = 0; r < 8; ++r) {
        const int m = mbase + r;
        if (m >= M) continue;
        float v = acc[mi][ni][r] + bsv;
        if (res) v += res[(size_t)((resDiv > 1) ? (m / resDiv) : m) * N + n];
        v = v * scv + shv;
        if (act == 1)      v = fmaxf(v, 0.f);
        else if (act == 2) { float u = v;
          float tt = 0.7978845608028654f * (u + 0.044715f * u * u * u);
          v = 0.5f * u * (1.f + tanhf(tt)); }
        else if (act == 3) v = (v >= 0.f) ? v : 0.2f * v;
        C[(size_t)m * ldc + colOff + n] = v;
      }
    }
}

// ---------------- helper kernels -------------------------------------------
__global__ void k_transpose3(const float* in, float* out, int b_, int n_) {
  int t = blockIdx.x * blockDim.x + threadIdx.x;
  if (t >= b_ * n_ * 3) return;
  int c = t % 3, n = (t / 3) % n_, b = t / (3 * n_);
  out[t] = in[(size_t)b * 3 * n_ + (size_t)c * n_ + n];
}

__global__ void k_stride3(const float* in, float* out, int b_, int nin, int ng, int stride) {
  int t = blockIdx.x * blockDim.x + threadIdx.x;
  if (t >= b_ * ng * 3) return;
  int c = t % 3, i = (t / 3) % ng, b = t / (3 * ng);
  out[t] = in[((size_t)b * nin + (size_t)i * stride) * 3 + c];
}

// wave-per-query k-nearest selection (ascending sq-distance, tie -> low idx)
__global__ __launch_bounds__(32) void k_topk(
    const float* q, const float* db, int nq, int ndb, int k,
    int* outIdx, float* outD, int flatBase)
{
  __shared__ unsigned chosen[128];          // up to 4096 candidates
  int qi = blockIdx.x, lane = threadIdx.x;
  int b = qi / nq;
  for (int i = lane; i < 128; i += 32) chosen[i] = 0u;
  __syncthreads();
  const float* Q = q + (size_t)qi * 3;
  float qx = Q[0], qy = Q[1], qz = Q[2];
  const float* DB = db + (size_t)b * ndb * 3;
  for (int sel = 0; sel < k; ++sel) {
    float bd = 3.4e38f; int bi = 0x7fffffff;
    for (int j = lane; j < ndb; j += 32) {
      if ((chosen[j >> 5] >> (j & 31)) & 1u) continue;
      float dx = DB[(size_t)j * 3 + 0] - qx;
      float dy = DB[(size_t)j * 3 + 1] - qy;
      float dz = DB[(size_t)j * 3 + 2] - qz;
      float d = dx * dx + dy * dy + dz * dz;
      if (d < bd || (d == bd && j < bi)) { bd = d; bi = j; }
    }
    for (int m = 16; m; m >>= 1) {
      float od = __shfl_xor(bd, m, 32);
      int   oi = __shfl_xor(bi, m, 32);
      if (od < bd || (od == bd && oi < bi)) { bd = od; bi = oi; }
    }
    if (lane == 0) {
      chosen[bi >> 5] |= 1u << (bi & 31);
      outIdx[(size_t)qi * k + sel] = bi + b * flatBase;
      if (outD) outD[(size_t)qi * k + sel] = bd;
    }
    __syncthreads();
  }
}

__global__ __launch_bounds__(128) void k_kpconv(
    const float* pts_t, const float* centers, const int* flat,
    const float* kp, const float* kpw, const float* bng, const float* bnb,
    float* out)
{
  __shared__ float wf[45];
  int q = blockIdx.x, tid = threadIdx.x;
  if (tid < 45) wf[tid] = 0.f;
  __syncthreads();
  int fi = flat[(size_t)q * 128 + tid];
  float ax = pts_t[(size_t)fi * 3 + 0];
  float ay = pts_t[(size_t)fi * 3 + 1];
  float az = pts_t[(size_t)fi * 3 + 2];
  float rx = ax - centers[(size_t)q * 3 + 0];
  float ry = ay - centers[(size_t)q * 3 + 1];
  float rz = az - centers[(size_t)q * 3 + 2];
  for (int k = 0; k < 15; ++k) {
    float dx = rx - kp[k * 3 + 0], dy = ry - kp[k * 3 + 1], dz = rz - kp[k * 3 + 2];
    float d = sqrtf(dx * dx + dy * dy + dz * dz);
    float w = 1.f - d * 25.f;                         // 1 - d/SIGMA, SIGMA=0.04
    if (w > 0.f) {
      atomicAdd(&wf[k * 3 + 0], w * ax);
      atomicAdd(&wf[k * 3 + 1], w * ay);
      atomicAdd(&wf[k * 3 + 2], w * az);
    }
  }
  __syncthreads();
  if (tid < 96) {
    float a = 0.f;
    for (int k = 0; k < 15; ++k)
      for (int c = 0; c < 3; ++c)
        a += wf[k * 3 + c] * kpw[((size_t)k * 6 + 3 + c) * 96 + tid];
    a = a * bng[tid] + bnb[tid];
    out[(size_t)q * 96 + tid] = (a >= 0.f) ? a : 0.2f * a;
  }
}

__global__ void k_add(float* x, const float* p, int n) {
  int t = blockIdx.x * blockDim.x + threadIdx.x;
  if (t < n) x[t] += p[t];
}

__global__ __launch_bounds__(32) void k_ln(
    const float* in, float* out, const float* g, const float* b, int rows, int C)
{
  int r = blockIdx.x, lane = threadIdx.x;
  const float* X = in + (size_t)r * C;
  float s = 0.f, s2 = 0.f;
  for (int c = lane; c < C; c += 32) { float v = X[c]; s += v; s2 += v * v; }
  for (int m = 16; m; m >>= 1) { s += __shfl_xor(s, m, 32); s2 += __shfl_xor(s2, m, 32); }
  float mean = s / C, var = s2 / C - mean * mean;
  float rs = rsqrtf(var + 1e-5f);
  float* O = out + (size_t)r * C;
  for (int c = lane; c < C; c += 32) O[c] = (X[c] - mean) * rs * g[c] + b[c];
}

// local-KNN attention: one wave per (b,g,head)
__global__ __launch_bounds__(32) void k_attn(
    const float* qkv, const int* knn, float* out, int G, int D, int hd, int kloc)
{
  __shared__ float sc[64];
  __shared__ int   sn[64];
  int bid = blockIdx.x;
  int h = bid % 6, g = (bid / 6) % G, b = bid / (6 * G);
  int lane = threadIdx.x;
  size_t tok = (size_t)b * G + g;
  const float* qp = qkv + (tok * 3 + 0) * D + (size_t)h * hd;
  float q0 = (lane < hd) ? qp[lane] : 0.f;
  float q1 = (lane + 32 < hd) ? qp[lane + 32] : 0.f;
  float scl = rsqrtf((float)hd);
  for (int j = lane; j < kloc; j += 32) sn[j] = knn[tok * kloc + j];
  __syncthreads();
  for (int j = 0; j < kloc; ++j) {
    size_t nt = (size_t)b * G + sn[j];
    const float* kp = qkv + (nt * 3 + 1) * D + (size_t)h * hd;
    float p = q0 * ((lane < hd) ? kp[lane] : 0.f) +
              q1 * ((lane + 32 < hd) ? kp[lane + 32] : 0.f);
    for (int m = 16; m; m >>= 1) p += __shfl_xor(p, m, 32);
    if (lane == 0) sc[j] = p * scl;
  }
  __syncthreads();
  float mx = -3.4e38f;
  for (int j = 0; j < kloc; ++j) mx = fmaxf(mx, sc[j]);
  float sum = 0.f;
  for (int j = 0; j < kloc; ++j) sum += expf(sc[j] - mx);
  float inv = 1.f / sum;
  float o0 = 0.f, o1 = 0.f;
  for (int j = 0; j < kloc; ++j) {
    float w = expf(sc[j] - mx) * inv;
    size_t nt = (size_t)b * G + sn[j];
    const float* vp = qkv + (nt * 3 + 2) * D + (size_t)h * hd;
    if (lane < hd)       o0 += w * vp[lane];
    if (lane + 32 < hd)  o1 += w * vp[lane + 32];
  }
  float* O = out + tok * D + (size_t)h * hd;
  if (lane < hd)      O[lane] = o0;
  if (lane + 32 < hd) O[lane + 32] = o1;
}

__global__ void k_gather(const float* in, const int* idx, float* out, int rows, int C) {
  int t = blockIdx.x * blockDim.x + threadIdx.x;
  if (t >= rows * C) return;
  int r = t / C, c = t % C;
  out[t] = in[(size_t)idx[r] * C + c];
}

__global__ void k_gmax(const float* in, float* out, int R, int k2, int C) {
  int t = blockIdx.x * blockDim.x + threadIdx.x;
  if (t >= R * C) return;
  int r = t / C, c = t % C;
  const float* p = in + (size_t)r * k2 * C + c;
  float m = -3.4e38f;
  for (int j = 0; j < k2; ++j) m = fmaxf(m, p[(size_t)j * C]);
  out[t] = m;
}

__global__ void k_catb(const float* gmx, const float* f, float* out, int rows, int k2, int C) {
  int t = blockIdx.x * blockDim.x + threadIdx.x;
  if (t >= rows * 2 * C) return;
  int c = t % (2 * C), r = t / (2 * C);
  out[t] = (c < C) ? gmx[(size_t)(r / k2) * C + c] : f[(size_t)r * C + (c - C)];
}

__global__ void k_interp(const float* pts_t, const float* f, const int* id3,
                         const float* d3, float* cin, int G, int Dd)
{
  int row = blockIdx.x * blockDim.x + threadIdx.x;
  if (row >= 2 * 4096) return;
  int b = row / 4096;
  float w0 = 1.f / fmaxf(d3[row * 3 + 0], 1e-10f);
  float w1 = 1.f / fmaxf(d3[row * 3 + 1], 1e-10f);
  float w2 = 1.f / fmaxf(d3[row * 3 + 2], 1e-10f);
  float ws = w0 + w1 + w2; w0 /= ws; w1 /= ws; w2 /= ws;
  float* O = cin + (size_t)row * (Dd + 3);
  O[0] = pts_t[(size_t)row * 3 + 0];
  O[1] = pts_t[(size_t)row * 3 + 1];
  O[2] = pts_t[(size_t)row * 3 + 2];
  const float* f0 = f + ((size_t)b * G + id3[row * 3 + 0]) * Dd;
  const float* f1 = f + ((size_t)b * G + id3[row * 3 + 1]) * Dd;
  const float* f2 = f + ((size_t)b * G + id3[row * 3 + 2]) * Dd;
  for (int c = 0; c < Dd; ++c) O[3 + c] = w0 * f0[c] + w1 * f1[c] + w2 * f2[c];
}

__global__ void k_maxmean(const float* x, float* g, int n_, int C) {
  int t = blockIdx.x * blockDim.x + threadIdx.x;
  if (t >= 2 * C) return;
  int b = t / C, c = t % C;
  const float* p = x + (size_t)b * n_ * C + c;
  float m = -3.4e38f, s = 0.f;
  for (int i = 0; i < n_; ++i) { float v = p[(size_t)i * C]; m = fmaxf(m, v); s += v; }
  g[t] = m + s / n_;
}

__global__ void k_gall(const float* gfeat, const float* cls, const float* lw,
                       const float* lbg, const float* lbb, float* gall) {
  int t = blockIdx.x * blockDim.x + threadIdx.x;
  if (t >= 2 * 3136) return;
  int b = t / 3136, c = t % 3136;
  float v;
  if (c < 3072) v = gfeat[b * 3072 + c];
  else {
    int o = c - 3072;
    float a = 0.f;
    for (int i = 0; i < 10; ++i) a += cls[b * 10 + i] * lw[i * 64 + o];
    a = a * lbg[o] + lbb[o];
    v = (a >= 0.f) ? a : 0.2f * a;
  }
  gall[t] = v;
}

__global__ void k_lsm(float* x, int rows, int C) {
  int r = blockIdx.x * blockDim.x + threadIdx.x;
  if (r >= rows) return;
  float* p = x + (size_t)r * C;
  float m = -3.4e38f;
  for (int c = 0; c < C; ++c) m = fmaxf(m, p[c]);
  float s = 0.f;
  for (int c = 0; c < C; ++c) s += expf(p[c] - m);
  float l = m + logf(s);
  for (int c = 0; c < C; ++c) p[c] -= l;
}

// ---------------- host-side helpers ----------------------------------------
static inline dim3 g1(long long n, int b = 256) { return dim3((unsigned)((n + b - 1) / b)); }

static inline void gemm(hipStream_t st, const float* A, const float* W, const float* bias,
                        const float* scale, const float* shift, const float* res, float* C,
                        int M, int N, int K, int ldc = -1, int colOff = 0,
                        int act = 0, int resDiv = 1)
{
  if (ldc < 0) ldc = N;
  dim3 grid((N + 63) / 64, (M + 127) / 128);
  k_gemm_wmma<<<grid, 256, 0, st>>>(A, W, bias, scale, shift, res, C,
                                    M, N, K, ldc, colOff, act, resDiv);
}

static void run_stage(hipStream_t st, void* const* d_in, int blkBase, int posBase,
                      float* x, const float* centers, int G, int D, int kloc,
                      float* posb, float* hbuf, float* qkvb, float* obuf,
                      float* t1, int* knn)
{
  auto P = [&](int i) { return (const float*)d_in[i]; };
  const int BG = 2 * G;
  // pos_embed: lin -> gelu -> lin
  gemm(st, centers, P(posBase + 1), P(posBase + 0), nullptr, nullptr, nullptr,
       hbuf, BG, D, 3, -1, 0, 2);
  gemm(st, hbuf, P(posBase + 3), P(posBase + 2), nullptr, nullptr, nullptr,
       posb, BG, D, D);
  k_topk<<<BG, 32, 0, st>>>(centers, centers, G, G, kloc, knn, nullptr, 0);
  for (int j = 0; j < 5; ++j) {
    int pb = blkBase + j * 12;
    const float *fc1b = P(pb + 0), *fc1w = P(pb + 1), *fc2b = P(pb + 2), *fc2w = P(pb + 3);
    const float *ln1b = P(pb + 4), *ln1g = P(pb + 5), *ln2b = P(pb + 6), *ln2g = P(pb + 7);
    const float *projb = P(pb + 8), *projw = P(pb + 9), *qkvbi = P(pb + 10), *qkvw = P(pb + 11);
    k_add<<<g1((long long)BG * D), 256, 0, st>>>(x, posb, BG * D);
    k_ln<<<BG, 32, 0, st>>>(x, hbuf, ln1g, ln1b, BG, D);
    gemm(st, hbuf, qkvw, qkvbi, nullptr, nullptr, nullptr, qkvb, BG, 3 * D, D);
    k_attn<<<BG * 6, 32, 0, st>>>(qkvb, knn, obuf, G, D, D / 6, kloc);
    gemm(st, obuf, projw, projb, nullptr, nullptr, x, x, BG, D, D);
    k_ln<<<BG, 32, 0, st>>>(x, hbuf, ln2g, ln2b, BG, D);
    gemm(st, hbuf, fc1w, fc1b, nullptr, nullptr, nullptr, t1, BG, 4 * D, D, -1, 0, 2);
    gemm(st, t1, fc2w, fc2b, nullptr, nullptr, x, x, BG, D, 4 * D);
  }
}

// ---------------------------------------------------------------------------
extern "C" void kernel_launch(void* const* d_in, const int* in_sizes, int n_in,
                              void* d_out, int out_size, void* d_ws, size_t ws_size,
                              hipStream_t stream)
{
  (void)in_sizes; (void)n_in; (void)out_size; (void)ws_size;
  const float* pts = (const float*)d_in[0];
  const float* cls = (const float*)d_in[1];
  auto P  = [&](int i) { return (const float*)d_in[i]; };

  // params leaf index table (JAX tree order: sorted dict keys, list order)
  const int BLK0 = 2, BLK1 = 62, BLK2 = 122;                  // blocks (12 leaves/block)
  const int BN1B = 182, BN1G = 183, BN2B = 184, BN2G = 185, BN3B = 186, BN3G = 187;
  const int C1B = 188, C1W = 189, C2B = 190, C2W = 191;
  const int C3B = 192, C3W = 193, C4B = 194, C4W = 195;
  const int KPTS = 196, KPBNB = 197, KPBNG = 198, KPW = 199;
  const int LBNB = 200, LBNG = 201, LW = 202;
  const int NRM = 203;                                        // lvl: b=NRM+2l, g=NRM+1+2l
  const int POS0 = 209;                                       // lvl stride 4: l1b,l1w,l2b,l2w
  const int PRP0 = 221;                                       // lvl stride 4
  const int TOK0 = 233;                                       // t stride 8: l1..l4 (b,w)

  // ---------------- workspace arena ----------------
  char* base = (char*)d_ws; size_t off = 0;
  auto af = [&](size_t ne) -> float* { float* p = (float*)(base + off);
             off += ((ne * 4 + 255) & ~(size_t)255); return p; };
  auto ai = [&](size_t ne) -> int* { int* p = (int*)(base + off);
             off += ((ne * 4 + 255) & ~(size_t)255); return p; };

  // persistent
  float* pts_t = af((size_t)2 * 4096 * 3);
  float* c0  = af((size_t)2 * 512 * 3);
  float* c1c = af((size_t)2 * 256 * 3);
  float* c2c = af((size_t)2 * 64 * 3);
  int* i0 = ai((size_t)2 * 512 * 128);
  int* i1 = ai((size_t)2 * 256 * 64);
  int* i2 = ai((size_t)2 * 64 * 64);
  float* f0 = af((size_t)2 * 512 * 96);
  float* f1 = af((size_t)2 * 256 * 192);
  float* f2 = af((size_t)2 * 64 * 384);
  float* xcat = af((size_t)2 * 4096 * 3072);
  // encoder scratch (max over stages)
  float* posb = af(98304);
  float* hbuf = af(98304);
  float* qkvb = af(294912);
  float* obuf = af(98304);
  float* t1b  = af(393216);
  int*   knn  = ai((size_t)2 * 512 * 64);
  const size_t scr = off;

  // ---------------- grouping / centers ----------------
  k_transpose3<<<g1((long long)2 * 4096 * 3), 256, 0, stream>>>(pts, pts_t, 2, 4096);
  k_stride3<<<g1((long long)2 * 512 * 3), 256, 0, stream>>>(pts_t, c0, 2, 4096, 512, 8);
  k_topk<<<2 * 512, 32, 0, stream>>>(c0, pts_t, 512, 4096, 128, i0, nullptr, 4096);
  k_stride3<<<g1((long long)2 * 256 * 3), 256, 0, stream>>>(c0, c1c, 2, 512, 256, 2);
  k_topk<<<2 * 256, 32, 0, stream>>>(c1c, c0, 256, 512, 64, i1, nullptr, 512);
  k_stride3<<<g1((long long)2 * 64 * 3), 256, 0, stream>>>(c1c, c2c, 2, 256, 64, 4);
  k_topk<<<2 * 64, 32, 0, stream>>>(c2c, c1c, 64, 256, 64, i2, nullptr, 256);

  // ---------------- stage 0: KPConv tokens + encoder ----------------
  k_kpconv<<<2 * 512, 128, 0, stream>>>(pts_t, c0, i0, P(KPTS), P(KPW),
                                        P(KPBNG), P(KPBNB), f0);
  run_stage(stream, d_in, BLK0, POS0, f0, c0, 512, 96, 16,
            posb, hbuf, qkvb, obuf, t1b, knn);

  // ---------------- token embed + encoder, levels 1 & 2 ----------------
  for (int lvl = 1; lvl <= 2; ++lvl) {
    const int g2  = (lvl == 1) ? 256 : 64;
    const int k2  = 64;
    const int Din = (lvl == 1) ? 96 : 192;
    const int Dout= (lvl == 1) ? 192 : 384;
    const int rows = 2 * g2 * k2;
    const int tb = TOK0 + (lvl - 1) * 8;
    const int* idx = (lvl == 1) ? i1 : i2;
    const float* xin = (lvl == 1) ? f0 : f1;
    float* xout = (lvl == 1) ? f1 : f2;

    off = scr;
    float* xn  = af((size_t)rows * Din);
    float* fb1 = af((size_t)rows * 128);
    float* fb2 = af((size_t)rows * 256);
    float* gmx = af((size_t)2 * g2 * 256);
    float* cat = af((size_t)rows * 512);
    float* f3  = af((size_t)rows * 512);
    float* f4  = af((size_t)rows * Dout);

    k_gather<<<g1((long long)rows * Din), 256, 0, stream>>>(xin, idx, xn, rows, Din);
    gemm(stream, xn, P(tb + 1), P(tb + 0), nullptr, nullptr, nullptr,
         fb1, rows, 128, Din, -1, 0, 1);
    gemm(stream, fb1, P(tb + 3), P(tb + 2), nullptr, nullptr, nullptr,
         fb2, rows, 256, 128);
    k_gmax<<<g1((long long)2 * g2 * 256), 256, 0, stream>>>(fb2, gmx, 2 * g2, k2, 256);
    k_catb<<<g1((long long)rows * 512), 256, 0, stream>>>(gmx, fb2, cat, rows, k2, 256);
    gemm(stream, cat, P(tb + 5), P(tb + 4), nullptr, nullptr, nullptr,
         f3, rows, 512, 512, -1, 0, 1);
    gemm(stream, f3, P(tb + 7), P(tb + 6), nullptr, nullptr, nullptr,
         f4, rows, Dout, 512);
    k_gmax<<<g1((long long)2 * g2 * Dout), 256, 0, stream>>>(f4, xout, 2 * g2, k2, Dout);

    if (lvl == 1)
      run_stage(stream, d_in, BLK1, POS0 + 4, f1, c1c, 256, 192, 32,
                posb, hbuf, qkvb, obuf, t1b, knn);
    else
      run_stage(stream, d_in, BLK2, POS0 + 8, f2, c2c, 64, 384, 64,
                posb, hbuf, qkvb, obuf, t1b, knn);
  }

  // ---------------- feature propagation into xcat (B,N,3072) ----------------
  {
    off = scr;
    int*   id3   = ai((size_t)2 * 4096 * 3);
    float* dd3   = af((size_t)2 * 4096 * 3);
    float* fnorm = af(98304);
    float* cin   = af((size_t)8192 * 387);
    float* pb    = af((size_t)8192 * 1536);
    const int   Gs[3] = {512, 256, 64};
    const int   Ds[3] = {96, 192, 384};
    const float* cents[3] = {c0, c1c, c2c};
    const float* feats[3] = {f0, f1, f2};
    for (int lvl = 0; lvl < 3; ++lvl) {
      int G = Gs[lvl], Dd = Ds[lvl];
      k_ln<<<2 * G, 32, 0, stream>>>(feats[lvl], fnorm, P(NRM + 1 + 2 * lvl),
                                     P(NRM + 2 * lvl), 2 * G, Dd);
      k_topk<<<2 * 4096, 32, 0, stream>>>(pts_t, cents[lvl], 4096, G, 3, id3, dd3, 0);
      k_interp<<<g1(8192), 256, 0, stream>>>(pts_t, fnorm, id3, dd3, cin, G, Dd);
      gemm(stream, cin, P(PRP0 + lvl * 4 + 1), P(PRP0 + lvl * 4 + 0),
           nullptr, nullptr, nullptr, pb, 8192, 1536, Dd + 3, -1, 0, 1);
      gemm(stream, pb, P(PRP0 + lvl * 4 + 3), P(PRP0 + lvl * 4 + 2),
           nullptr, nullptr, nullptr, xcat, 8192, 1024, 1536, 3072, lvl * 1024, 1);
    }
  }

  // ---------------- segmentation head ----------------
  {
    off = scr;
    float* gfeat = af((size_t)2 * 3072);
    float* gall  = af((size_t)2 * 3136);
    float* g1b   = af((size_t)2 * 1024);
    float* c1o   = af((size_t)8192 * 1024);
    float* c2o   = af((size_t)8192 * 512);
    float* c3o   = af((size_t)8192 * 256);

    k_maxmean<<<g1((long long)2 * 3072), 256, 0, stream>>>(xcat, gfeat, 4096, 3072);
    k_gall<<<g1((long long)2 * 3136), 256, 0, stream>>>(gfeat, cls, P(LW),
                                                        P(LBNG), P(LBNB), gall);
    // c1 split: gall @ W[0:3136] (+bias) per batch, then xcat @ W[3136:] + bcast residual
    gemm(stream, gall, P(C1W), P(C1B), nullptr, nullptr, nullptr,
         g1b, 2, 1024, 3136);
    gemm(stream, xcat, P(C1W) + (size_t)3136 * 1024, nullptr,
         P(BN1G), P(BN1B), g1b, c1o, 8192, 1024, 3072, -1, 0, 1, 4096);
    gemm(stream, c1o, P(C2W), P(C2B), P(BN2G), P(BN2B), nullptr,
         c2o, 8192, 512, 1024, -1, 0, 1);
    gemm(stream, c2o, P(C3W), P(C3B), P(BN3G), P(BN3B), nullptr,
         c3o, 8192, 256, 512, -1, 0, 1);
    gemm(stream, c3o, P(C4W), P(C4B), nullptr, nullptr, nullptr,
         (float*)d_out, 8192, 50, 256);
    k_lsm<<<g1(8192), 256, 0, stream>>>((float*)d_out, 8192, 50);
  }
}
